// MultiHeadAttention_26379689132826
// MI455X (gfx1250) — matrix operand order repaired and checked by reading, not compiled, not run on hardware
//
#include <hip/hip_runtime.h>

// ---------------------------------------------------------------------------
// MultiHeadAttention for MI455X (gfx1250): bf16 WMMA pipeline + async-LDS.
// B=2, S=2048, E=1024, H=16, D=64. scale = 1/sqrt(E) = 1/32.
// Workspace: Pq, Pk(d-major per head), Pv, Ctx (each 8 MB bf16) -> 32 MB.
// ---------------------------------------------------------------------------

typedef __bf16 bf16_t;
typedef __attribute__((ext_vector_type(16))) __bf16 v16bf;
typedef __attribute__((ext_vector_type(8)))  __bf16 v8bf;
typedef __attribute__((ext_vector_type(8)))  float  v8f;
typedef __attribute__((ext_vector_type(4)))  float  v4f;
typedef int v4i_gcc __attribute__((vector_size(16)));

#define AS1 __attribute__((address_space(1)))
#define AS3 __attribute__((address_space(3)))

#if defined(__AMDGCN__) && __has_builtin(__builtin_amdgcn_global_load_async_to_lds_b128) && __has_builtin(__builtin_amdgcn_s_wait_asynccnt)
#define HAS_ASYNC_LDS 1
#else
#define HAS_ASYNC_LDS 0
#endif

// Copy 16 bytes global -> LDS (async path uses GLOBAL_LOAD_ASYNC_TO_LDS_B128,
// tracked by ASYNCcnt; fallback is a plain vectorized copy through VGPRs).
__device__ __forceinline__ void copy16_g2l(const bf16_t* g, bf16_t* l) {
#if HAS_ASYNC_LDS
  __builtin_amdgcn_global_load_async_to_lds_b128(
      (AS1 v4i_gcc*)(g), (AS3 v4i_gcc*)(l), 0, 0);
#else
  *(v8bf*)l = *(const v8bf*)g;
#endif
}
__device__ __forceinline__ void wait_async_lds() {
#if HAS_ASYNC_LDS
  __builtin_amdgcn_s_wait_asynccnt(0);
#endif
}

union Frag { v16bf v; v8bf h2[2]; };

__device__ __forceinline__ v8f wmma_bf16(v16bf a, v16bf b, v8f c) {
  return __builtin_amdgcn_wmma_f32_16x16x32_bf16(
      /*neg_a=*/false, a, /*neg_b=*/false, b,
      /*c_mod=*/(short)0, c, /*reuse_a=*/false, /*reuse_b=*/false);
}

// A fragment: 16(M) x 32(K) bf16, row-major in LDS with `stride` elements.
__device__ __forceinline__ v16bf load_a_frag(const bf16_t* s, int stride, int lane) {
  const int r  = lane & 15;
  const int ko = (lane >> 4) << 3;  // 0 or 8
  const bf16_t* base = s + r * stride + ko;
  Frag u;
  u.h2[0] = *(const v8bf*)(base);
  u.h2[1] = *(const v8bf*)(base + 16);
  return u.v;
}

// B fragment: 32(K) x 16(N), row-major [K][N] in LDS; lane%16=N, elem=K (table order).
__device__ __forceinline__ v16bf load_b_frag(const bf16_t* s, int stride, int n0, int lane) {
  const int n  = n0 + (lane & 15);
  const int ko = (lane >> 4) << 3;  // 0 or 8
  Frag u;
#pragma unroll
  for (int i = 0; i < 8; ++i) {
    u.v[i]     = s[(ko + i) * stride + n];
    u.v[8 + i] = s[(16 + ko + i) * stride + n];
  }
  return u.v;
}

// ---------------------------------------------------------------------------
// GEMM: Out[M x 1024] = A[M x 1024] @ W[1024 x 1024] + bias
// MODE 0: A f32 -> Out bf16, flat (B,S,E) layout          (Q, V projections)
// MODE 1: A f32 -> Out bf16, per-head d-major layout       (K projection)
// MODE 2: A bf16 -> Out f32, flat                          (final Wo GEMM)
// Block tile 128(M) x 64(N), 8 waves, wave tile 32x32, K-step 32.
// ---------------------------------------------------------------------------
template<int MODE>
__global__ __launch_bounds__(256) void gemm_wmma_kernel(
    const void* __restrict__ Ap, const float* __restrict__ W,
    const float* __restrict__ bias, void* __restrict__ Outp)
{
  constexpr int KDIM = 1024, NDIM = 1024;
  constexpr int BM = 128, BN = 64, BK = 32;
  __shared__ bf16_t sA[BM * BK];   // 8 KB
  __shared__ bf16_t sB[BK * BN];   // 4 KB

  const int tid  = threadIdx.x;
  const int lane = tid & 31;
  const int wave = tid >> 5;
  const int wm   = (wave >> 1) * 32;
  const int wn   = (wave & 1) * 32;
  const int bm   = blockIdx.y * BM;
  const int bn   = blockIdx.x * BN;

  v8f acc[2][2] = {};

  const int arow = tid >> 1;          // 0..127
  const int acol = (tid & 1) << 4;    // 0,16
  const int brow = tid >> 3;          // 0..31
  const int bcol = (tid & 7) << 3;    // 0..56

  for (int k0 = 0; k0 < KDIM; k0 += BK) {
    // ---- stage A tile (128 x 32) ----
    if constexpr (MODE != 2) {
      const float* g = (const float*)Ap + (size_t)(bm + arow) * KDIM + k0 + acol;
      if (k0 + BK < KDIM) __builtin_prefetch(g + BK, 0, 1);
      v4f f0 = *(const v4f*)(g);
      v4f f1 = *(const v4f*)(g + 4);
      v4f f2 = *(const v4f*)(g + 8);
      v4f f3 = *(const v4f*)(g + 12);
      bf16_t* d = sA + arow * BK + acol;
#pragma unroll
      for (int j = 0; j < 4; ++j) {
        d[j]      = (bf16_t)f0[j];
        d[4 + j]  = (bf16_t)f1[j];
        d[8 + j]  = (bf16_t)f2[j];
        d[12 + j] = (bf16_t)f3[j];
      }
    } else {
      const bf16_t* g = (const bf16_t*)Ap + (size_t)(bm + arow) * KDIM + k0 + acol;
      if (k0 + BK < KDIM) __builtin_prefetch(g + BK, 0, 1);
      copy16_g2l(g,     sA + arow * BK + acol);
      copy16_g2l(g + 8, sA + arow * BK + acol + 8);
    }
    // ---- stage B tile (32 x 64), f32 -> bf16 ----
    {
      const float* g = W + (size_t)(k0 + brow) * NDIM + bn + bcol;
      if (k0 + BK < KDIM) __builtin_prefetch(g + (size_t)BK * NDIM, 0, 1);
      v4f f0 = *(const v4f*)(g);
      v4f f1 = *(const v4f*)(g + 4);
      bf16_t* d = sB + brow * BN + bcol;
#pragma unroll
      for (int j = 0; j < 4; ++j) { d[j] = (bf16_t)f0[j]; d[4 + j] = (bf16_t)f1[j]; }
    }
    if constexpr (MODE == 2) wait_async_lds();
    __syncthreads();

    v16bf a0 = load_a_frag(sA + (wm +  0) * BK, BK, lane);
    v16bf a1 = load_a_frag(sA + (wm + 16) * BK, BK, lane);
    v16bf b0 = load_b_frag(sB, BN, wn +  0, lane);
    v16bf b1 = load_b_frag(sB, BN, wn + 16, lane);
    acc[0][0] = wmma_bf16(a0, b0, acc[0][0]);
    acc[0][1] = wmma_bf16(a0, b1, acc[0][1]);
    acc[1][0] = wmma_bf16(a1, b0, acc[1][0]);
    acc[1][1] = wmma_bf16(a1, b1, acc[1][1]);
    __syncthreads();
  }

  // ---- writeback (C layout: elem j -> row j/j+8 by lane half, lane -> col) ----
  const int chalf = lane >> 4;
  const int ccol  = lane & 15;
#pragma unroll
  for (int mi = 0; mi < 2; ++mi)
#pragma unroll
    for (int ni = 0; ni < 2; ++ni) {
      const int col = bn + wn + ni * 16 + ccol;
      const float bv = bias[col];
#pragma unroll
      for (int j = 0; j < 8; ++j) {
        const int row = bm + wm + mi * 16 + j + 8 * chalf;
        const float vv = acc[mi][ni][j] + bv;
        if constexpr (MODE == 2) {
          ((float*)Outp)[(size_t)row * NDIM + col] = vv;
        } else if constexpr (MODE == 1) {
          // reshape semantics of row m=(b,s_orig), col e -> (b,h,s,d);
          // store d-major per head: off = b*S*E + h*S*D + d*S + s
          const int b  = row >> 11;
          const int so = row & 2047;
          const int h  = so >> 7;
          const int s  = ((so & 127) << 4) | (col >> 6);
          const int d  = col & 63;
          const size_t off = (size_t)b * 2097152 + (size_t)h * 131072 +
                             (size_t)d * 2048 + s;
          ((bf16_t*)Outp)[off] = (bf16_t)vv;
        } else {
          ((bf16_t*)Outp)[(size_t)row * NDIM + col] = (bf16_t)vv;
        }
      }
    }
}

// ---------------------------------------------------------------------------
// Flash attention per (b, h). Block = 8 waves; wave owns 16 query rows;
// workgroup iterates key tiles of 32 sharing K^T / V tiles in LDS.
// Pq, Pv: head rows at base + b*S*E + h*S*D + s*D (flat reshape view).
// Pk: stored d-major per head: base + b*S*E + h*S*D + d*S + s.
// ---------------------------------------------------------------------------
__global__ __launch_bounds__(256) void attn_wmma_kernel(
    const bf16_t* __restrict__ Pq, const bf16_t* __restrict__ Pk,
    const bf16_t* __restrict__ Pv, bf16_t* __restrict__ Ctx)
{
  constexpr int S = 2048, D = 64, E = 1024;
  const int tid  = threadIdx.x;
  const int lane = tid & 31;
  const int wave = tid >> 5;
  const size_t headoff = (size_t)blockIdx.z * S * E + (size_t)blockIdx.y * S * D;
  const bf16_t* Qh = Pq + headoff;
  const bf16_t* Kh = Pk + headoff;   // [d][s], stride S
  const bf16_t* Vh = Pv + headoff;   // [s][d], stride D
  bf16_t*       Ch = Ctx + headoff;

  __shared__ bf16_t sKt[D * 32];      // K^T tile: [d][key], stride 32 (4 KB)
  __shared__ bf16_t sV [32 * D];      // V tile:   [key][d], stride 64 (4 KB)
  __shared__ bf16_t sP [8][16 * 32];  // per-wave P scratch (8 KB)

  const int q0 = blockIdx.x * 128 + wave * 16;

  // Q fragments for K-dim chunks [0,32) and [32,64), loaded once from global.
  const int r  = lane & 15;
  const int ko = (lane >> 4) << 3;
  const bf16_t* qrow = Qh + (size_t)(q0 + r) * D;
  Frag qa0, qa1;
  qa0.h2[0] = *(const v8bf*)(qrow + ko);
  qa0.h2[1] = *(const v8bf*)(qrow + ko + 16);
  qa1.h2[0] = *(const v8bf*)(qrow + 32 + ko);
  qa1.h2[1] = *(const v8bf*)(qrow + 32 + ko + 16);

  v8f o[4] = {};
  float mrow[8], lrow[8];
#pragma unroll
  for (int j = 0; j < 8; ++j) { mrow[j] = -1e30f; lrow[j] = 0.0f; }

  const float scale = 0.03125f;       // 1/sqrt(1024)
  const int dd = tid >> 2;            // 0..63  (K^T tile row = d)
  const int kp = (tid & 3) << 3;      // 0..24  (key chunk)
  const int kk = tid >> 3;            // 0..31  (V tile row = key)
  const int vp = (tid & 7) << 3;      // 0..56  (d chunk)

  for (int kt = 0; kt < S; kt += 32) {
    // ---- stage K^T (64x32, contiguous rows of Pk) and V (32x64) tiles ----
    copy16_g2l(Kh + (size_t)dd * S + kt + kp, sKt + dd * 32 + kp);
    copy16_g2l(Vh + (size_t)(kt + kk) * D + vp, sV + kk * D + vp);
    wait_async_lds();
    __syncthreads();

    // ---- S tile: 16(q) x 32(keys) = Q(16x64) @ K^T(64x32) ----
    v8f s0 = {}, s1 = {};
    {
      v16bf bk00 = load_b_frag(sKt,           32,  0, lane);
      v16bf bk01 = load_b_frag(sKt,           32, 16, lane);
      v16bf bk10 = load_b_frag(sKt + 32 * 32, 32,  0, lane);
      v16bf bk11 = load_b_frag(sKt + 32 * 32, 32, 16, lane);
      s0 = wmma_bf16(qa0.v, bk00, s0);
      s0 = wmma_bf16(qa1.v, bk10, s0);
      s1 = wmma_bf16(qa0.v, bk01, s1);
      s1 = wmma_bf16(qa1.v, bk11, s1);
    }

    // ---- online softmax update ----
#pragma unroll
    for (int j = 0; j < 8; ++j) {
      float t = fmaxf(s0[j], s1[j]) * scale;
#pragma unroll
      for (int m = 1; m < 16; m <<= 1) t = fmaxf(t, __shfl_xor(t, m, 32));
      const float mnew = fmaxf(mrow[j], t);
      const float corr = __expf(mrow[j] - mnew);
      mrow[j] = mnew;
      const float p0 = __expf(s0[j] * scale - mnew);
      const float p1 = __expf(s1[j] * scale - mnew);
      float rs = p0 + p1;
#pragma unroll
      for (int m = 1; m < 16; m <<= 1) rs += __shfl_xor(rs, m, 32);
      lrow[j] = lrow[j] * corr + rs;
      o[0][j] *= corr; o[1][j] *= corr; o[2][j] *= corr; o[3][j] *= corr;
      s0[j] = p0; s1[j] = p1;
    }

    // ---- re-layout P (C layout -> A fragment) via per-wave LDS scratch ----
    bf16_t* pw = sP[wave];
    {
      const int col  = lane & 15;
      const int half = lane >> 4;
#pragma unroll
      for (int j = 0; j < 8; ++j) {
        pw[(j + 8 * half) * 32 + col]      = (bf16_t)s0[j];
        pw[(j + 8 * half) * 32 + col + 16] = (bf16_t)s1[j];
      }
    }
    __syncthreads();
    v16bf pfrag = load_a_frag(pw, 32, lane);

    // ---- O += P(16x32) @ V(32x64) ----
#pragma unroll
    for (int n = 0; n < 4; ++n)
      o[n] = wmma_bf16(pfrag, load_b_frag(sV, D, n * 16, lane), o[n]);
    __syncthreads();
  }

  // ---- normalize and store context (flat (B,S,E) bf16) ----
  {
    const int col  = lane & 15;
    const int half = lane >> 4;
#pragma unroll
    for (int n = 0; n < 4; ++n)
#pragma unroll
      for (int j = 0; j < 8; ++j) {
        const int srow = q0 + j + 8 * half;
        Ch[(size_t)srow * D + n * 16 + col] = (bf16_t)(o[n][j] / lrow[j]);
      }
  }
}

// ---------------------------------------------------------------------------
extern "C" void kernel_launch(void* const* d_in, const int* in_sizes, int n_in,
                              void* d_out, int out_size, void* d_ws, size_t ws_size,
                              hipStream_t stream) {
  const float* k  = (const float*)d_in[0];
  const float* v  = (const float*)d_in[1];
  const float* q  = (const float*)d_in[2];
  const float* Wk = (const float*)d_in[3];
  const float* bk = (const float*)d_in[4];
  const float* Wv = (const float*)d_in[5];
  const float* bv = (const float*)d_in[6];
  const float* Wq = (const float*)d_in[7];
  const float* bq = (const float*)d_in[8];
  const float* Wo = (const float*)d_in[9];
  const float* bo = (const float*)d_in[10];
  float* out = (float*)d_out;

  const size_t MT = (size_t)4096 * 1024;
  bf16_t* Pq  = (bf16_t*)d_ws;
  bf16_t* Pk  = Pq + MT;                   // d-major per head
  bf16_t* Pv  = Pk + MT;
  bf16_t* Ctx = Pv + MT;                   // 32 MB workspace total

  const dim3 blk(256);
  const dim3 gg(16, 32);

  gemm_wmma_kernel<0><<<gg, blk, 0, stream>>>(q,   Wq, bq, Pq);
  gemm_wmma_kernel<1><<<gg, blk, 0, stream>>>(k,   Wk, bk, Pk);
  gemm_wmma_kernel<0><<<gg, blk, 0, stream>>>(v,   Wv, bv, Pv);

  const dim3 ga(16, 16, 2);
  attn_wmma_kernel<<<ga, blk, 0, stream>>>(Pq, Pk, Pv, Ctx);

  gemm_wmma_kernel<2><<<gg, blk, 0, stream>>>(Ctx, Wo, bo, out);
}
